// dct_channel_block_50044958933487
// MI455X (gfx1250) — matrix operand order repaired and checked by reading, not compiled, not run on hardware
//
#include <hip/hip_runtime.h>

// ---------------------------------------------------------------------------
// Fused DCT(96) -> LayerNorm -> Linear(96->192)+ReLU -> Linear(192->96)+sigmoid
// -> LayerNorm -> x * weight     for MI455X (gfx1250, wave32, WMMA f16)
// ---------------------------------------------------------------------------

typedef _Float16 v16h __attribute__((ext_vector_type(16)));
typedef _Float16 v8h  __attribute__((ext_vector_type(8)));
typedef float    v8f  __attribute__((ext_vector_type(8)));
typedef float    v4f  __attribute__((ext_vector_type(4)));

#define NF   96      // feature dim
#define NH   192     // hidden dim
#define WAVES 8      // waves per block
#define ROWS_PER_WAVE 16
#define LN_EPS 1e-6f

#define WMMA_F16(a, b, c) \
  __builtin_amdgcn_wmma_f32_16x16x32_f16(false, (a), false, (b), (short)0, (c), false, false)

// ds_swizzle_b32 group-of-32 XOR patterns: offset = (xor<<10) | (or<<5) | and
#define SWZ_ADD(x, patt) \
  ((x) + __int_as_float(__builtin_amdgcn_ds_swizzle(__float_as_int(x), (patt))))

union Frag16 { v16h v; v8h h2[2]; _Float16 h[16]; };
union Acc8   { v8f v; float f[8]; };

// Butterfly sum across the 16 lanes of each half of a wave32 (XOR masks <= 8
// never cross the 16-lane boundary, matching the WMMA C-layout row split).
__device__ __forceinline__ float half16_reduce(float x) {
  x = SWZ_ADD(x, 0x041F);   // SWAPX1
  x = SWZ_ADD(x, 0x081F);   // SWAPX2
  x = SWZ_ADD(x, 0x101F);   // SWAPX4
  x = SWZ_ADD(x, 0x201F);   // SWAPX8
  return x;
}

extern "C" __global__ __launch_bounds__(WAVES * 32)
void fused_dct_mlp_ln_gfx1250(const float* __restrict__ x,
                              const float* __restrict__ W1,   // [192][96]
                              const float* __restrict__ W2,   // [96][192]
                              const float* __restrict__ gamma,
                              const float* __restrict__ beta,
                              float* __restrict__ out,
                              long totalRows) {
  extern __shared__ _Float16 smem[];
  _Float16* Dlds  = smem;                       // [96][96]  DCT matrix, row-major [n][k]
  _Float16* W1lds = Dlds  + NF * NF;            // [192][96] B2tr[n][k] = W1[n][k]
  _Float16* W2lds = W1lds + NH * NF;            // [96][192] B3tr[n][h] = W2[n][h]
  _Float16* stage = W2lds + NF * NH;            // [WAVES][16][192] per-wave staging

  const int tid = threadIdx.x;

  // ---- cooperative weight staging (f32 -> f16, DCT computed on the fly) ----
  const float PI = 3.14159265358979323846f;
  for (int idx = tid; idx < NF * NF; idx += blockDim.x) {
    int k = idx / NF, i = idx - k * NF;
    float v = 2.0f * __cosf(PI * (2.0f * (float)i + 1.0f) * (float)k * (1.0f / (2.0f * NF)));
    Dlds[idx] = (_Float16)v;
  }
  for (int idx = tid; idx < NH * NF; idx += blockDim.x) W1lds[idx] = (_Float16)W1[idx];
  for (int idx = tid; idx < NF * NH; idx += blockDim.x) W2lds[idx] = (_Float16)W2[idx];
  __syncthreads();

  const int wave = tid >> 5;
  const int lane = tid & 31;
  const int half = lane >> 4;      // which 16-lane half
  const int ln   = lane & 15;      // lane within half (column / row index)
  const long rowBase = ((long)blockIdx.x * WAVES + wave) * ROWS_PER_WAVE;
  if (rowBase + ROWS_PER_WAVE > totalRows) return;   // wave-uniform guard

  _Float16* st = stage + wave * (ROWS_PER_WAVE * NH);

  // Per-lane LayerNorm affine params, one per 16-wide feature tile.
  float g[6], bt[6];
#pragma unroll
  for (int nt = 0; nt < 6; ++nt) { g[nt] = gamma[nt * 16 + ln]; bt[nt] = beta[nt * 16 + ln]; }

  // ---- Load x tile [16,96] as f16 A-fragments (3 K-chunks of 32) ----------
  // 16-bit A layout: lane<16 -> K0+[0,8) and K0+16+[0,8); lane>=16 the rest.
  Frag16 ax[3];
  {
    const float* xr = x + (rowBase + ln) * NF;
#pragma unroll
    for (int kf = 0; kf < 3; ++kf) {
      const int K0 = kf * 32;
      v4f lo0 = *(const v4f*)(xr + K0 + half * 8);
      v4f lo1 = *(const v4f*)(xr + K0 + half * 8 + 4);
      v4f hi0 = *(const v4f*)(xr + K0 + 16 + half * 8);
      v4f hi1 = *(const v4f*)(xr + K0 + 16 + half * 8 + 4);
#pragma unroll
      for (int j = 0; j < 4; ++j) {
        ax[kf].h[j]      = (_Float16)lo0[j];
        ax[kf].h[4 + j]  = (_Float16)lo1[j];
        ax[kf].h[8 + j]  = (_Float16)hi0[j];
        ax[kf].h[12 + j] = (_Float16)hi1[j];
      }
    }
  }

  // ---- Stage 1: DCT  y = x * D^T  (B[k][n] = D[n][k], stored row-major) ---
  Acc8 acc[6];
#pragma unroll
  for (int nt = 0; nt < 6; ++nt) {
    v8f c = {};
#pragma unroll
    for (int kf = 0; kf < 3; ++kf) {
      Frag16 b;
      const _Float16* bp = Dlds + (nt * 16 + ln) * NF + kf * 32 + half * 8;
      b.h2[0] = *(const v8h*)bp;
      b.h2[1] = *(const v8h*)(bp + 16);
      c = WMMA_F16(ax[kf].v, b.v, c);
    }
    acc[nt].v = c;
  }

  // ---- LayerNorm #1 (per row over 96 features) ----------------------------
  float mu[8], rs[8];
#pragma unroll
  for (int r = 0; r < 8; ++r) {
    float s1 = 0.f, s2 = 0.f;
#pragma unroll
    for (int nt = 0; nt < 6; ++nt) { float v = acc[nt].f[r]; s1 += v; s2 += v * v; }
    s1 = half16_reduce(s1);
    s2 = half16_reduce(s2);
    float m   = s1 * (1.0f / NF);
    float var = s2 * (1.0f / NF) - m * m;
    mu[r] = m;
    rs[r] = rsqrtf(var + LN_EPS);
  }
  // Write normalized tile to staging as f16 [16][96] (C layout -> row major).
#pragma unroll
  for (int nt = 0; nt < 6; ++nt)
#pragma unroll
    for (int r = 0; r < 8; ++r) {
      float v = (acc[nt].f[r] - mu[r]) * rs[r] * g[nt] + bt[nt];
      st[(r + half * 8) * NF + nt * 16 + ln] = (_Float16)v;
    }

  // ---- Stage 2: h = relu(lr @ W1^T), 96 -> 192 ----------------------------
  Frag16 a2[3];
#pragma unroll
  for (int kf = 0; kf < 3; ++kf) {
    const _Float16* p = st + ln * NF + kf * 32;
    a2[kf].h2[0] = *(const v8h*)(p + half * 8);
    a2[kf].h2[1] = *(const v8h*)(p + 16 + half * 8);
  }
#pragma unroll
  for (int nt = 0; nt < 12; ++nt) {
    v8f c = {};
#pragma unroll
    for (int kf = 0; kf < 3; ++kf) {
      Frag16 b;
      const _Float16* bp = W1lds + (nt * 16 + ln) * NF + kf * 32 + half * 8;
      b.h2[0] = *(const v8h*)bp;
      b.h2[1] = *(const v8h*)(bp + 16);
      c = WMMA_F16(a2[kf].v, b.v, c);
    }
    Acc8 h; h.v = c;
#pragma unroll
    for (int r = 0; r < 8; ++r) {
      float v = fmaxf(h.f[r], 0.0f);
      st[(r + half * 8) * NH + nt * 16 + ln] = (_Float16)v;   // [16][192]
    }
  }

  // ---- Stage 3: s = sigmoid(h @ W2^T), 192 -> 96 --------------------------
  Frag16 a3[6];
#pragma unroll
  for (int kf = 0; kf < 6; ++kf) {
    const _Float16* p = st + ln * NH + kf * 32;
    a3[kf].h2[0] = *(const v8h*)(p + half * 8);
    a3[kf].h2[1] = *(const v8h*)(p + 16 + half * 8);
  }
#pragma unroll
  for (int nt = 0; nt < 6; ++nt) {
    v8f c = {};
#pragma unroll
    for (int kf = 0; kf < 6; ++kf) {
      Frag16 b;
      const _Float16* bp = W2lds + (nt * 16 + ln) * NH + kf * 32 + half * 8;
      b.h2[0] = *(const v8h*)bp;
      b.h2[1] = *(const v8h*)(bp + 16);
      c = WMMA_F16(a3[kf].v, b.v, c);
    }
    Acc8 s; s.v = c;
#pragma unroll
    for (int r = 0; r < 8; ++r)
      acc[nt].f[r] = 1.0f / (1.0f + __expf(-s.f[r]));
  }

  // ---- LayerNorm #2 + out = x * weight ------------------------------------
#pragma unroll
  for (int r = 0; r < 8; ++r) {
    float s1 = 0.f, s2 = 0.f;
#pragma unroll
    for (int nt = 0; nt < 6; ++nt) { float v = acc[nt].f[r]; s1 += v; s2 += v * v; }
    s1 = half16_reduce(s1);
    s2 = half16_reduce(s2);
    float m   = s1 * (1.0f / NF);
    float var = s2 * (1.0f / NF) - m * m;
    mu[r] = m;
    rs[r] = rsqrtf(var + LN_EPS);
  }
#pragma unroll
  for (int nt = 0; nt < 6; ++nt)
#pragma unroll
    for (int r = 0; r < 8; ++r) {
      float w = (acc[nt].f[r] - mu[r]) * rs[r] * g[nt] + bt[nt];
      long idx = (rowBase + r + half * 8) * NF + nt * 16 + ln;
      out[idx] = x[idx] * w;      // coalesced across the 16 lanes of each half
    }
}

extern "C" void kernel_launch(void* const* d_in, const int* in_sizes, int n_in,
                              void* d_out, int out_size, void* d_ws, size_t ws_size,
                              hipStream_t stream) {
  const float* x     = (const float*)d_in[0];
  const float* W1    = (const float*)d_in[1];
  const float* W2    = (const float*)d_in[2];
  const float* gamma = (const float*)d_in[3];
  const float* beta  = (const float*)d_in[4];
  float* out = (float*)d_out;

  const long totalRows = (long)in_sizes[0] / NF;              // 512*1024 = 524288
  const int  rowsPerBlock = WAVES * ROWS_PER_WAVE;            // 128
  const long nBlocks = (totalRows + rowsPerBlock - 1) / rowsPerBlock;

  const int smemBytes =
      (NF * NF + NH * NF + NF * NH + WAVES * ROWS_PER_WAVE * NH) * (int)sizeof(_Float16);

  // Raise dynamic-LDS cap (141 KB < 320 KB CDNA5 workgroup limit).
  // Idempotent + deterministic; not a stream operation.
  (void)hipFuncSetAttribute((const void*)fused_dct_mlp_ln_gfx1250,
                            hipFuncAttributeMaxDynamicSharedMemorySize, smemBytes);

  fused_dct_mlp_ln_gfx1250<<<dim3((unsigned)nBlocks), dim3(WAVES * 32), smemBytes, stream>>>(
      x, W1, W2, gamma, beta, out, totalRows);
}
